// GraphRNN_27101243638401
// MI455X (gfx1250) — matrix operand order repaired and checked by reading, not compiled, not run on hardware
//
#include <hip/hip_runtime.h>
#include <hip/hip_bf16.h>

typedef __attribute__((ext_vector_type(16))) __bf16 v16bf;
typedef __attribute__((ext_vector_type(8)))  float  v8f;

union FragU { uint4 u[2]; v16bf v; };

static constexpr int NN   = 10000;
static constexpr int TT   = 12;
static constexpr int DD   = 512;
static constexpr int HH   = 512;
static constexpr int EE   = 320000;
static constexpr int DIN  = 1024;   // D + H
static constexpr int MPAD = 10240;  // NN rounded up to 256 (block M tile)

// ---------------- hs = [x_t, state]  (f32 for scatter, bf16 for WMMA) -------
__global__ void build_hs_kernel(const float* __restrict__ xt, long xt_stride,
                                const float* __restrict__ state,
                                float* __restrict__ hs_f32,
                                __hip_bfloat16* __restrict__ hs_bf16) {
    int i = blockIdx.x;
    #pragma unroll
    for (int k = 0; k < 4; ++k) {
        int j = threadIdx.x + k * 256;
        float v = (j < DD) ? xt[(size_t)i * xt_stride + j]
                           : state[(size_t)i * HH + (j - DD)];
        size_t o = (size_t)i * DIN + j;
        hs_f32[o]  = v;
        hs_bf16[o] = __float2bfloat16(v);
    }
}

// ---------------- segment sum: agg[dst] += h[src] ---------------------------
__global__ void scatter_add_kernel(const float* __restrict__ h,
                                   const int* __restrict__ src,
                                   const int* __restrict__ dst,
                                   float* __restrict__ agg) {
    int e = blockIdx.x;
    int c = threadIdx.x;                       // 256 float4 chunks == DIN
    int s = src[e], d = dst[e];
    float4 v = reinterpret_cast<const float4*>(h + (size_t)s * DIN)[c];
    float* p = agg + (size_t)d * DIN + c * 4;
    atomicAdd(p + 0, v.x);
    atomicAdd(p + 1, v.y);
    atomicAdd(p + 2, v.z);
    atomicAdd(p + 3, v.w);
}

// ---------------- agg_bf16 = bf16(agg * inv_deg[i]) -------------------------
__global__ void scale_bf16_kernel(const float* __restrict__ agg,
                                  const float* __restrict__ invdeg,
                                  __hip_bfloat16* __restrict__ out) {
    int i = blockIdx.x;
    float s = invdeg[i];
    #pragma unroll
    for (int k = 0; k < 4; ++k) {
        int j = threadIdx.x + k * 256;
        size_t o = (size_t)i * DIN + j;
        out[o] = __float2bfloat16(agg[o] * s);
    }
}

// ---------------- WMMA fragment loads (ISA §7.12.2 layouts) -----------------
__device__ __forceinline__ v16bf load_a_frag(const __hip_bfloat16* A, int K,
                                             int m, int kb, int lane) {
    int kb0 = kb + ((lane & 16) ? 8 : 0);      // lanes 16-31 carry K+8 group
    const __hip_bfloat16* p = A + (size_t)m * K + kb0;
    FragU f;
    f.u[0] = *reinterpret_cast<const uint4*>(p);        // K = kb0 .. kb0+7
    f.u[1] = *reinterpret_cast<const uint4*>(p + 16);   // K = kb0+16 .. +23
    return f.v;
}
__device__ __forceinline__ v16bf load_b_frag(const __hip_bfloat16* Wt, int K,
                                             int n, int kb, int lane) {
    int kbB = kb + ((lane & 16) ? 8 : 0);      // lanes 16-31 carry K+8 group
    const uint4* p = reinterpret_cast<const uint4*>(Wt + (size_t)n * K + kbB);
    FragU f;
    f.u[0] = p[0];
    f.u[1] = p[2];                              // K = kbB+16 .. +23 (Wt is OxK)
    return f.v;
}

// --------- C = A1@W1t + A2@W2t + bias (bf16 in, f32 acc) --------------------
// 8 waves arranged 4(M)x2(N); wave tile 64Mx64N = 16 WMMA accumulators.
// A rows must be readable up to MPAD (padded activation buffers).
__global__ __launch_bounds__(256)
void gemm_dual_kernel(const __hip_bfloat16* __restrict__ A1,
                      const __hip_bfloat16* __restrict__ W1,
                      const __hip_bfloat16* __restrict__ A2,
                      const __hip_bfloat16* __restrict__ W2,
                      const float* __restrict__ bias,
                      float* __restrict__ C,
                      int M, int K, int O) {
    int lane = threadIdx.x & 31;
    int wave = threadIdx.x >> 5;
    int wm = wave >> 1;                         // 0..3
    int wn = wave & 1;                          // 0..1
    int m0 = blockIdx.x * 256 + wm * 64;
    int n0 = blockIdx.y * 128 + wn * 64;
    if (m0 >= M) return;                        // wave-uniform guard
    int mr = m0 + (lane & 15);

    v8f acc[4][4];                              // [m-tile][n-tile]
    #pragma unroll
    for (int t = 0; t < 4; ++t) {
        float bv = bias[n0 + t * 16 + (lane & 15)];
        #pragma unroll
        for (int i = 0; i < 4; ++i)
            #pragma unroll
            for (int r = 0; r < 8; ++r) acc[i][t][r] = bv;
    }

    for (int kb = 0; kb < K; kb += 32) {
        v16bf a[4];
        #pragma unroll
        for (int i = 0; i < 4; ++i)
            a[i] = load_a_frag(A1, K, mr + i * 16, kb, lane);
        #pragma unroll
        for (int t = 0; t < 4; ++t) {
            v16bf b = load_b_frag(W1, K, n0 + t * 16 + (lane & 15), kb, lane);
            #pragma unroll
            for (int i = 0; i < 4; ++i)
                acc[i][t] = __builtin_amdgcn_wmma_f32_16x16x32_bf16(
                    false, a[i], false, b, (short)0, acc[i][t], false, false);
        }
    }
    for (int kb = 0; kb < K; kb += 32) {
        v16bf a[4];
        #pragma unroll
        for (int i = 0; i < 4; ++i)
            a[i] = load_a_frag(A2, K, mr + i * 16, kb, lane);
        #pragma unroll
        for (int t = 0; t < 4; ++t) {
            v16bf b = load_b_frag(W2, K, n0 + t * 16 + (lane & 15), kb, lane);
            #pragma unroll
            for (int i = 0; i < 4; ++i)
                acc[i][t] = __builtin_amdgcn_wmma_f32_16x16x32_bf16(
                    false, a[i], false, b, (short)0, acc[i][t], false, false);
        }
    }

    int roff = ((lane & 16) ? 8 : 0);           // C/D layout: VGPR r -> M=r(+8)
    #pragma unroll
    for (int i = 0; i < 4; ++i) {
        int mi = m0 + i * 16;
        if (mi >= M) break;                     // uniform; tiles are full (M%16==0)
        #pragma unroll
        for (int t = 0; t < 4; ++t) {
            int col = n0 + t * 16 + (lane & 15);
            #pragma unroll
            for (int r = 0; r < 8; ++r)
                C[(size_t)(mi + roff + r) * O + col] = acc[i][t][r];
        }
    }
}

// ---------------- sigmoid gates, build cin = [x_t, r*state] -----------------
__global__ void gate_kernel(const float* __restrict__ G,
                            const float* __restrict__ state,
                            const float* __restrict__ xt, long xt_stride,
                            float* __restrict__ cin_f32,
                            __hip_bfloat16* __restrict__ cin_bf16,
                            float* __restrict__ u_buf) {
    int i = blockIdx.x;
    #pragma unroll
    for (int k = 0; k < 4; ++k) {
        int j = threadIdx.x + k * 128;
        float r = 1.f / (1.f + expf(-G[(size_t)i * DIN + j]));
        float u = 1.f / (1.f + expf(-G[(size_t)i * DIN + HH + j]));
        float xv = xt[(size_t)i * xt_stride + j];
        float rs = r * state[(size_t)i * HH + j];
        size_t o = (size_t)i * DIN;
        cin_f32[o + j]       = xv;
        cin_f32[o + HH + j]  = rs;
        cin_bf16[o + j]      = __float2bfloat16(xv);
        cin_bf16[o + HH + j] = __float2bfloat16(rs);
        u_buf[(size_t)i * HH + j] = u;
    }
}

// ---------------- GRU update: state = u*state + (1-u)*tanh(Cc) --------------
__global__ void update_kernel(const float* __restrict__ Cc,
                              const float* __restrict__ u_buf,
                              float* __restrict__ state,
                              float* __restrict__ seq_out,    // relu'd, or null
                              float* __restrict__ final_out)  // raw, or null
{
    int i = blockIdx.x;
    #pragma unroll
    for (int k = 0; k < 4; ++k) {
        int j = threadIdx.x + k * 128;
        size_t o = (size_t)i * HH + j;
        float c  = tanhf(Cc[o]);
        float u  = u_buf[o];
        float ns = u * state[o] + (1.f - u) * c;
        state[o] = ns;
        if (seq_out)   seq_out[o]   = fmaxf(ns, 0.f);
        if (final_out) final_out[o] = ns;
    }
}

// ---------------- weight fp32 (KxO) -> bf16 transposed (OxK) ----------------
__global__ void convert_weight_kernel(const float* __restrict__ W,
                                      __hip_bfloat16* __restrict__ Wt,
                                      int K, int O) {
    int o = blockIdx.x;
    for (int k = threadIdx.x; k < K; k += blockDim.x)
        Wt[(size_t)o * K + k] = __float2bfloat16(W[(size_t)k * O + o]);
}

__global__ void combine_bias_kernel(const float* __restrict__ a,
                                    const float* __restrict__ b,
                                    float* __restrict__ out, int n) {
    int i = blockIdx.x * blockDim.x + threadIdx.x;
    if (i < n) out[i] = a[i] + b[i];
}

__global__ void deg_kernel(const int* __restrict__ dst, float* __restrict__ deg) {
    int e = blockIdx.x * blockDim.x + threadIdx.x;
    if (e < EE) atomicAdd(&deg[dst[e]], 1.0f);
}
__global__ void invdeg_kernel(const float* __restrict__ deg, float* __restrict__ inv) {
    int i = blockIdx.x * blockDim.x + threadIdx.x;
    if (i < NN) inv[i] = 1.0f / fmaxf(deg[i], 1.0f);
}

// ---------------- logits = out @ fc_w + fc_b (wave-per-node) ----------------
__global__ void logits_kernel(const float* __restrict__ out,
                              const float* __restrict__ fc_w,
                              const float* __restrict__ fc_b,
                              float* __restrict__ logits) {
    int lane = threadIdx.x & 31;
    int wave = threadIdx.x >> 5;
    int node = blockIdx.x * 4 + wave;
    if (node >= NN) return;
    float s = 0.f;
    for (int k = lane; k < HH; k += 32)
        s += out[(size_t)node * HH + k] * fc_w[k];
    #pragma unroll
    for (int off = 16; off > 0; off >>= 1)
        s += __shfl_xor(s, off, 32);
    if (lane == 0) logits[node] = s + fc_b[0];
}

extern "C" void kernel_launch(void* const* d_in, const int* in_sizes, int n_in,
                              void* d_out, int out_size, void* d_ws, size_t ws_size,
                              hipStream_t stream) {
    const float* inputs    = (const float*)d_in[0];
    const int*   src       = (const int*)d_in[1];
    const int*   dst       = (const int*)d_in[2];
    const float* Wg_self   = (const float*)d_in[3];
    const float* Wg_neigh  = (const float*)d_in[4];
    const float* bg        = (const float*)d_in[5];
    const float* gate_bias = (const float*)d_in[6];
    const float* Wc_self   = (const float*)d_in[7];
    const float* Wc_neigh  = (const float*)d_in[8];
    const float* bc        = (const float*)d_in[9];
    const float* cand_bias = (const float*)d_in[10];
    const float* fc_w      = (const float*)d_in[11];
    const float* fc_b      = (const float*)d_in[12];

    float* out_logits = (float*)d_out;           // [N]
    float* out_feat   = (float*)d_out + NN;      // [N, H]

    size_t off = 0;
    auto carve = [&](size_t bytes) -> void* {
        void* p = (char*)d_ws + off;
        off += (bytes + 255) & ~(size_t)255;
        return p;
    };
    float* xseq    = (float*)carve((size_t)TT * NN * HH * 4);  // inter-layer seq
    float* state   = (float*)carve((size_t)NN * HH * 4);
    float* hs_f32  = (float*)carve((size_t)NN * DIN * 4);      // aliases cin_f32
    __hip_bfloat16* hs_bf16 = (__hip_bfloat16*)carve((size_t)MPAD * DIN * 2);
    float* agg_f32 = (float*)carve((size_t)NN * DIN * 4);
    __hip_bfloat16* agg_bf16 = (__hip_bfloat16*)carve((size_t)MPAD * DIN * 2);
    float* Gbuf    = (float*)carve((size_t)NN * DIN * 4);      // gate preact / Cc
    float* u_buf   = (float*)carve((size_t)NN * HH * 4);
    float* degb    = (float*)carve((size_t)NN * 4);
    float* invdeg  = (float*)carve((size_t)NN * 4);
    __hip_bfloat16* Wgs_t = (__hip_bfloat16*)carve((size_t)DIN * DIN * 2);
    __hip_bfloat16* Wgn_t = (__hip_bfloat16*)carve((size_t)DIN * DIN * 2);
    __hip_bfloat16* Wcs_t = (__hip_bfloat16*)carve((size_t)HH * DIN * 2);
    __hip_bfloat16* Wcn_t = (__hip_bfloat16*)carve((size_t)HH * DIN * 2);
    float* bias_g  = (float*)carve(DIN * 4);
    float* bias_c  = (float*)carve(HH * 4);

    // degree / inverse degree (once)
    hipMemsetAsync(degb, 0, NN * 4, stream);
    deg_kernel<<<(EE + 255) / 256, 256, 0, stream>>>(dst, degb);
    invdeg_kernel<<<(NN + 255) / 256, 256, 0, stream>>>(degb, invdeg);

    dim3 gemmGridG((NN + 255) / 256, DIN / 128);  // 40 x 8
    dim3 gemmGridC((NN + 255) / 256, HH / 128);   // 40 x 4

    for (int l = 0; l < 2; ++l) {
        convert_weight_kernel<<<DIN, 256, 0, stream>>>(Wg_self  + (size_t)l * DIN * DIN, Wgs_t, DIN, DIN);
        convert_weight_kernel<<<DIN, 256, 0, stream>>>(Wg_neigh + (size_t)l * DIN * DIN, Wgn_t, DIN, DIN);
        convert_weight_kernel<<<HH,  256, 0, stream>>>(Wc_self  + (size_t)l * DIN * HH,  Wcs_t, DIN, HH);
        convert_weight_kernel<<<HH,  256, 0, stream>>>(Wc_neigh + (size_t)l * DIN * HH,  Wcn_t, DIN, HH);
        combine_bias_kernel<<<(DIN + 255) / 256, 256, 0, stream>>>(
            bg + (size_t)l * DIN, gate_bias + (size_t)l * DIN, bias_g, DIN);
        combine_bias_kernel<<<(HH + 255) / 256, 256, 0, stream>>>(
            bc + (size_t)l * HH, cand_bias + (size_t)l * HH, bias_c, HH);
        hipMemsetAsync(state, 0, (size_t)NN * HH * 4, stream);

        for (int t = 0; t < TT; ++t) {
            const float* xt; long xt_stride;
            if (l == 0) { xt = inputs + (size_t)t * DD;       xt_stride = (long)TT * DD; }
            else        { xt = xseq + (size_t)t * NN * HH;    xt_stride = HH; }

            build_hs_kernel<<<NN, 256, 0, stream>>>(xt, xt_stride, state, hs_f32, hs_bf16);

            hipMemsetAsync(agg_f32, 0, (size_t)NN * DIN * 4, stream);
            scatter_add_kernel<<<EE, 256, 0, stream>>>(hs_f32, src, dst, agg_f32);
            scale_bf16_kernel<<<NN, 256, 0, stream>>>(agg_f32, invdeg, agg_bf16);

            gemm_dual_kernel<<<gemmGridG, 256, 0, stream>>>(
                hs_bf16, Wgs_t, agg_bf16, Wgn_t, bias_g, Gbuf, NN, DIN, DIN);

            gate_kernel<<<NN, 128, 0, stream>>>(Gbuf, state, xt, xt_stride,
                                                hs_f32, hs_bf16, u_buf);

            hipMemsetAsync(agg_f32, 0, (size_t)NN * DIN * 4, stream);
            scatter_add_kernel<<<EE, 256, 0, stream>>>(hs_f32, src, dst, agg_f32);
            scale_bf16_kernel<<<NN, 256, 0, stream>>>(agg_f32, invdeg, agg_bf16);

            gemm_dual_kernel<<<gemmGridC, 256, 0, stream>>>(
                hs_bf16, Wcs_t, agg_bf16, Wcn_t, bias_c, Gbuf, NN, DIN, HH);

            float* seq_out = (l == 0) ? (xseq + (size_t)t * NN * HH) : nullptr;
            float* fin = (l == 1 && t == TT - 1) ? out_feat : nullptr;
            update_kernel<<<NN, 128, 0, stream>>>(Gbuf, u_buf, state, seq_out, fin);
        }
    }
    logits_kernel<<<(NN + 3) / 4, 128, 0, stream>>>(out_feat, fc_w, fc_b, out_logits);
}
